// SoftPAGCrossAttnProcessor_23725399343123
// MI455X (gfx1250) — matrix operand order repaired and checked by reading, not compiled, not run on hardware
//
#include <hip/hip_runtime.h>
#include <hip/hip_bf16.h>

// CDNA5 / gfx1250: wave32, WMMA 16x16x32 f16 -> f32, async LDS staging.
typedef __attribute__((ext_vector_type(16))) _Float16 v16h;
typedef __attribute__((ext_vector_type(8)))  float    v8f;
typedef __attribute__((ext_vector_type(4)))  unsigned u32x4;
typedef __attribute__((ext_vector_type(2)))  unsigned u32x2;
typedef __attribute__((ext_vector_type(4)))  float    f32x4;

#define NB  2
#define LSEQ 2048
#define DMODEL 1024
#define NH  16
#define DH  64

union Frag16 { v16h v; u32x4 u[2]; };

// Async global->LDS 16B copy per lane (ASYNCcnt-tracked).
// ldsAddr: DS byte address (low 32 bits of generic shared pointer).
__device__ inline void async_copy_b128(unsigned ldsAddr, const void* gaddr) {
  unsigned long long ga = (unsigned long long)gaddr;
  asm volatile("global_load_async_to_lds_b128 %0, %1, off"
               :: "v"(ldsAddr), "v"(ga) : "memory");
}
__device__ inline void wait_asynccnt0() {
  asm volatile("s_wait_asynccnt 0x0" ::: "memory");
}
__device__ inline unsigned lds_addr(const void* p) {
  return (unsigned)(unsigned long long)p;
}

// ---------------------------------------------------------------------------
// Kernel 1: fused Q/K/V projection.  C = X @ W  (f32 in, f16 out, split heads)
// Block tile 128x64, wave tile 32x32 (4 WMMA / K-step).
// grid = (DMODEL/64, (NB*LSEQ)/128, 3), block = 256 (8 waves)
// ---------------------------------------------------------------------------
__global__ __launch_bounds__(256)
void qkv_proj_kernel(const float* __restrict__ hidden,
                     const float* __restrict__ enc,
                     const float* __restrict__ Wq,
                     const float* __restrict__ Wk,
                     const float* __restrict__ Wv,
                     _Float16* __restrict__ Qh,
                     _Float16* __restrict__ Kh,
                     _Float16* __restrict__ Vh) {
  __shared__ _Float16 As[128][40];   // [m][k] tile, padded stride
  __shared__ _Float16 Bs[64][40];    // [n][k] tile (W transposed)

  const int tid  = threadIdx.x;
  const int wv   = tid >> 5;
  const int lane = tid & 31;
  const int ln16 = lane & 15;
  const int lh   = lane >> 4;
  const int wr   = wv & 3;        // wave row block (32 rows)
  const int wc   = wv >> 2;       // wave col block (32 cols)
  const int m0   = blockIdx.y * 128;
  const int n0   = blockIdx.x * 64;

  const float* X; const float* W; _Float16* O;
  if (blockIdx.z == 0)      { X = hidden; W = Wq; O = Qh; }
  else if (blockIdx.z == 1) { X = enc;    W = Wk; O = Kh; }
  else                      { X = enc;    W = Wv; O = Vh; }

  v8f acc[2][2];
  for (int at = 0; at < 2; ++at)
    for (int ct = 0; ct < 2; ++ct)
      for (int e = 0; e < 8; ++e) acc[at][ct][e] = 0.f;

#pragma unroll 1
  for (int k0 = 0; k0 < DMODEL; k0 += 32) {
    // stage A: 128 rows x 32 cols f32 -> f16
    for (int rr = 0; rr < 4; ++rr) {
      int row = (tid >> 3) + rr * 32;
      int c4  = (tid & 7) * 4;
      f32x4 vsrc = *(const f32x4*)(X + (size_t)(m0 + row) * DMODEL + k0 + c4);
      As[row][c4 + 0] = (_Float16)vsrc.x;
      As[row][c4 + 1] = (_Float16)vsrc.y;
      As[row][c4 + 2] = (_Float16)vsrc.z;
      As[row][c4 + 3] = (_Float16)vsrc.w;
    }
    // stage B transposed: Bs[n][k] = W[k0+k][n0+n]
    {
      int n  = tid & 63;
      int kb = (tid >> 6) * 8;
      for (int i = 0; i < 8; ++i)
        Bs[n][kb + i] = (_Float16)W[(size_t)(k0 + kb + i) * DMODEL + n0 + n];
    }
    if (k0 + 32 < DMODEL)
      __builtin_prefetch(X + (size_t)(m0 + (tid >> 1)) * DMODEL + k0 + 32, 0, 1);
    __syncthreads();

    Frag16 af[2], bf[2];
    for (int at = 0; at < 2; ++at) {
      int am = wr * 32 + at * 16 + ln16;
      af[at].u[0] = *(const u32x4*)(&As[am][lh * 8]);
      af[at].u[1] = *(const u32x4*)(&As[am][16 + lh * 8]);
    }
    for (int ct = 0; ct < 2; ++ct) {
      int bn = wc * 32 + ct * 16 + ln16;
      bf[ct].u[0] = *(const u32x4*)(&Bs[bn][lh * 8]);
      bf[ct].u[1] = *(const u32x4*)(&Bs[bn][16 + lh * 8]);
    }
    for (int at = 0; at < 2; ++at)
      for (int ct = 0; ct < 2; ++ct)
        acc[at][ct] = __builtin_amdgcn_wmma_f32_16x16x32_f16(
            false, af[at].v, false, bf[ct].v, (short)0, acc[at][ct], false, false);
    __syncthreads();
  }

  // epilogue: split heads -> [b][h][l][dh] f16
  for (int at = 0; at < 2; ++at) {
    for (int ct = 0; ct < 2; ++ct) {
      for (int r = 0; r < 8; ++r) {
        int row  = m0 + wr * 32 + at * 16 + r + 8 * lh;
        int col  = n0 + wc * 32 + ct * 16 + ln16;
        int bi   = row >> 11;           // / LSEQ
        int li   = row & (LSEQ - 1);
        int head = col >> 6;
        int dc   = col & (DH - 1);
        O[(((size_t)(bi * NH + head) * LSEQ) + li) * DH + dc] =
            (_Float16)acc[at][ct][r];
      }
    }
  }
}

// ---------------------------------------------------------------------------
// Kernel 2: flash attention + SoftPAG blend.
// out_row = 0.5 * softmax(QK^T/8) @ V  + 0.5 * V[q]   (Lq==Lk -> eye diagonal)
// grid = (LSEQ/128, NB*NH), block = 256 (8 waves, 16 query rows each)
// K tile staged via GLOBAL_LOAD_ASYNC_TO_LDS_B128 (ASYNCcnt).
// ---------------------------------------------------------------------------
__global__ __launch_bounds__(256)
void flash_kernel(const _Float16* __restrict__ Qh,
                  const _Float16* __restrict__ Kh,
                  const _Float16* __restrict__ Vh,
                  _Float16* __restrict__ Ob) {
  __shared__ _Float16 Ks[64][72];       // key tile   [key][d]
  __shared__ _Float16 Vt[64][72];       // V tile transposed [d][key]
  __shared__ _Float16 Ps[8][16][72];    // per-wave P staging [m][key]

  const int tid  = threadIdx.x;
  const int wv   = tid >> 5;
  const int lane = tid & 31;
  const int ln16 = lane & 15;
  const int lh   = lane >> 4;
  const int bh   = blockIdx.y;        // b*NH + head
  const int bi   = bh >> 4;
  const int head = bh & 15;

  const _Float16* Qbase = Qh + (size_t)bh * LSEQ * DH;
  const _Float16* Kbase = Kh + (size_t)bh * LSEQ * DH;
  const _Float16* Vbase = Vh + (size_t)bh * LSEQ * DH;

  const int q0   = blockIdx.x * 128;
  const int qrow = q0 + wv * 16 + ln16;      // A-matrix row this lane serves

  // Q fragments: D=64 -> two K=32 chunks, kept resident.
  v16h qf[2];
  {
    const _Float16* qr = Qbase + (size_t)qrow * DH;
    for (int fr = 0; fr < 2; ++fr) {
      Frag16 f;
      f.u[0] = *(const u32x4*)(qr + fr * 32 + lh * 8);
      f.u[1] = *(const u32x4*)(qr + fr * 32 + 16 + lh * 8);
      qf[fr] = f.v;
    }
  }

  float m_row[8], l_row[8];
  v8f o[4];
  for (int r = 0; r < 8; ++r) { m_row[r] = -1e30f; l_row[r] = 0.f; }
  for (int ct = 0; ct < 4; ++ct)
    for (int e = 0; e < 8; ++e) o[ct][e] = 0.f;

  const float scale = 0.125f;   // 1/sqrt(64)

#pragma unroll 1
  for (int kt = 0; kt < LSEQ / 64; ++kt) {
    const int k0 = kt * 64;
    // stage K tile via async global->LDS (64x64 halfs = 512 x 16B)
    for (int it = 0; it < 2; ++it) {
      int idx = tid + it * 256;
      int row = idx >> 3;
      int c8  = (idx & 7) * 8;
      async_copy_b128(lds_addr(&Ks[row][c8]),
                      Kbase + (size_t)(k0 + row) * DH + c8);
    }
    // stage V tile transposed: Vt[d][key]  (needs 16-bit transpose -> scalar)
    {
      int key   = tid >> 2;
      int dbase = (tid & 3) * 16;
      const _Float16* vr = Vbase + (size_t)(k0 + key) * DH + dbase;
      for (int i = 0; i < 16; ++i) Vt[dbase + i][key] = vr[i];
    }
    // prefetch next tiles into cache while this one is consumed
    if (kt + 1 < LSEQ / 64) {
      __builtin_prefetch(Kbase + (size_t)(k0 + 64 + (tid >> 2)) * DH, 0, 1);
      __builtin_prefetch(Vbase + (size_t)(k0 + 64 + (tid >> 2)) * DH, 0, 1);
    }
    wait_asynccnt0();
    __syncthreads();

    // S = Q @ K^T  (16 x 64 per wave, 4 C tiles)
    v8f s[4];
    for (int ct = 0; ct < 4; ++ct) {
      v8f sa; for (int e = 0; e < 8; ++e) sa[e] = 0.f;
      for (int fr = 0; fr < 2; ++fr) {
        Frag16 kf;
        kf.u[0] = *(const u32x4*)(&Ks[ct * 16 + ln16][fr * 32 + lh * 8]);
        kf.u[1] = *(const u32x4*)(&Ks[ct * 16 + ln16][fr * 32 + 16 + lh * 8]);
        sa = __builtin_amdgcn_wmma_f32_16x16x32_f16(
            false, qf[fr], false, kf.v, (short)0, sa, false, false);
      }
      s[ct] = sa;
    }

    // online softmax per row (row m = r + 8*lh; reduce across lanes sharing lh)
    float alpha[8];
    for (int r = 0; r < 8; ++r) {
      float mx = s[0][r];
      for (int ct = 1; ct < 4; ++ct) mx = fmaxf(mx, s[ct][r]);
      mx *= scale;
      for (int off = 1; off < 16; off <<= 1) mx = fmaxf(mx, __shfl_xor(mx, off, 32));
      float mn = fmaxf(m_row[r], mx);
      alpha[r] = __expf(m_row[r] - mn);
      m_row[r] = mn;
      float rs = 0.f;
      for (int ct = 0; ct < 4; ++ct) {
        float p = __expf(s[ct][r] * scale - mn);
        rs += p;
        Ps[wv][r + 8 * lh][ct * 16 + ln16] = (_Float16)p;
      }
      for (int off = 1; off < 16; off <<= 1) rs += __shfl_xor(rs, off, 32);
      l_row[r] = l_row[r] * alpha[r] + rs;
    }
    for (int ct = 0; ct < 4; ++ct)
      for (int r = 0; r < 8; ++r) o[ct][r] *= alpha[r];

    // P back from LDS in A layout (wave-private region; DS in-order per wave)
    v16h pf[2];
    for (int fr = 0; fr < 2; ++fr) {
      Frag16 f;
      f.u[0] = *(const u32x4*)(&Ps[wv][ln16][fr * 32 + lh * 8]);
      f.u[1] = *(const u32x4*)(&Ps[wv][ln16][fr * 32 + 16 + lh * 8]);
      pf[fr] = f.v;
    }
    // O += P @ V
    for (int ct = 0; ct < 4; ++ct) {
      for (int fr = 0; fr < 2; ++fr) {
        Frag16 vf;
        vf.u[0] = *(const u32x4*)(&Vt[ct * 16 + ln16][fr * 32 + lh * 8]);
        vf.u[1] = *(const u32x4*)(&Vt[ct * 16 + ln16][fr * 32 + 16 + lh * 8]);
        o[ct] = __builtin_amdgcn_wmma_f32_16x16x32_f16(
            false, pf[fr], false, vf.v, (short)0, o[ct], false, false);
      }
    }
    __syncthreads();
  }

  // epilogue: normalize, SoftPAG blend with V diagonal, store [b][l][h*DH+d]
  for (int ct = 0; ct < 4; ++ct) {
    for (int r = 0; r < 8; ++r) {
      int q  = q0 + wv * 16 + r + 8 * lh;
      int dc = ct * 16 + ln16;
      float vdiag = (float)Vbase[(size_t)q * DH + dc];
      float val = 0.5f * (o[ct][r] / l_row[r]) + 0.5f * vdiag;
      Ob[((size_t)(bi * LSEQ + q)) * DMODEL + head * DH + dc] = (_Float16)val;
    }
  }
}

// ---------------------------------------------------------------------------
// Kernel 3: output projection.  out = A @ Wo + bo  (f16 A, f32 out)
// Block tile 128x64, wave tile 32x32 (4 WMMA / K-step).
// grid = (DMODEL/64, (NB*LSEQ)/128), block = 256
// ---------------------------------------------------------------------------
__global__ __launch_bounds__(256)
void oproj_kernel(const _Float16* __restrict__ Ah,
                  const float* __restrict__ Wo,
                  const float* __restrict__ bo,
                  float* __restrict__ out) {
  __shared__ _Float16 As[128][40];
  __shared__ _Float16 Bs[64][40];

  const int tid  = threadIdx.x;
  const int wv   = tid >> 5;
  const int lane = tid & 31;
  const int ln16 = lane & 15;
  const int lh   = lane >> 4;
  const int wr   = wv & 3;
  const int wc   = wv >> 2;
  const int m0   = blockIdx.y * 128;
  const int n0   = blockIdx.x * 64;

  v8f acc[2][2];
  for (int at = 0; at < 2; ++at)
    for (int ct = 0; ct < 2; ++ct)
      for (int e = 0; e < 8; ++e) acc[at][ct][e] = 0.f;

#pragma unroll 1
  for (int k0 = 0; k0 < DMODEL; k0 += 32) {
    for (int rr = 0; rr < 4; ++rr) {
      int row = (tid >> 3) + rr * 32;
      int c4  = (tid & 7) * 4;
      *(u32x2*)(&As[row][c4]) =
          *(const u32x2*)(Ah + (size_t)(m0 + row) * DMODEL + k0 + c4);
    }
    {
      int n  = tid & 63;
      int kb = (tid >> 6) * 8;
      for (int i = 0; i < 8; ++i)
        Bs[n][kb + i] = (_Float16)Wo[(size_t)(k0 + kb + i) * DMODEL + n0 + n];
    }
    __syncthreads();

    Frag16 af[2], bf[2];
    for (int at = 0; at < 2; ++at) {
      int am = wr * 32 + at * 16 + ln16;
      af[at].u[0] = *(const u32x4*)(&As[am][lh * 8]);
      af[at].u[1] = *(const u32x4*)(&As[am][16 + lh * 8]);
    }
    for (int ct = 0; ct < 2; ++ct) {
      int bn = wc * 32 + ct * 16 + ln16;
      bf[ct].u[0] = *(const u32x4*)(&Bs[bn][lh * 8]);
      bf[ct].u[1] = *(const u32x4*)(&Bs[bn][16 + lh * 8]);
    }
    for (int at = 0; at < 2; ++at)
      for (int ct = 0; ct < 2; ++ct)
        acc[at][ct] = __builtin_amdgcn_wmma_f32_16x16x32_f16(
            false, af[at].v, false, bf[ct].v, (short)0, acc[at][ct], false, false);
    __syncthreads();
  }

  for (int at = 0; at < 2; ++at) {
    for (int ct = 0; ct < 2; ++ct) {
      for (int r = 0; r < 8; ++r) {
        int row = m0 + wr * 32 + at * 16 + r + 8 * lh;
        int col = n0 + wc * 32 + ct * 16 + ln16;
        out[(size_t)row * DMODEL + col] = acc[at][ct][r] + bo[col];
      }
    }
  }
}

// ---------------------------------------------------------------------------
extern "C" void kernel_launch(void* const* d_in, const int* in_sizes, int n_in,
                              void* d_out, int out_size, void* d_ws, size_t ws_size,
                              hipStream_t stream) {
  const float* hidden = (const float*)d_in[0];
  const float* enc    = (const float*)d_in[1];
  const float* Wq     = (const float*)d_in[2];
  const float* Wk     = (const float*)d_in[3];
  const float* Wv     = (const float*)d_in[4];
  const float* Wo     = (const float*)d_in[5];
  const float* bo     = (const float*)d_in[6];
  float*       out    = (float*)d_out;

  const size_t perTensor = (size_t)NB * NH * LSEQ * DH;   // 4,194,304 halfs
  _Float16* Qh = (_Float16*)d_ws;
  _Float16* Kh = Qh + perTensor;
  _Float16* Vh = Kh + perTensor;
  _Float16* Ab = Vh + perTensor;   // [b][l][h*DH+d] = NB*LSEQ*DMODEL halfs

  dim3 blk(256, 1, 1);
  dim3 gProj(DMODEL / 64, (NB * LSEQ) / 128, 3);
  qkv_proj_kernel<<<gProj, blk, 0, stream>>>(hidden, enc, Wq, Wk, Wv, Qh, Kh, Vh);

  dim3 gAttn(LSEQ / 128, NB * NH, 1);
  flash_kernel<<<gAttn, blk, 0, stream>>>(Qh, Kh, Vh, Ab);

  dim3 gO(DMODEL / 64, (NB * LSEQ) / 128, 1);
  oproj_kernel<<<gO, blk, 0, stream>>>(Ab, Wo, bo, out);
}
